// HeteroGraphSAGE_16612933501407
// MI455X (gfx1250) — compile-verified, matrix operand order checked by reading, not run on hardware
//
#include <hip/hip_runtime.h>
#include <hip/hip_bf16.h>
#include <stdint.h>

typedef __attribute__((ext_vector_type(16))) _Float16 v16h;
typedef __attribute__((ext_vector_type(8)))  _Float16 v8h;
typedef __attribute__((ext_vector_type(8)))  float    v8f;

#define NR   16384   // review nodes
#define FD   256     // feature / hidden dim
#define KNBR 16      // neighbor fan-out
#define BM   128
#define BN   128

enum { EPI_LSTM = 0, EPI_SAGE_ST = 1, EPI_SAGE_ADD = 2, EPI_GELU16 = 3 };

// Double-buffered A tile (async global->LDS), plus gate buffer for LSTM cell.
struct SmemPlain { _Float16 As[2][BM * 32]; };
struct SmemLSTM  { _Float16 As[2][BM * 32]; float gbuf[BM * 64]; };
template<int E> struct SmemSel { using T = SmemPlain; };
template<>      struct SmemSel<EPI_LSTM> { using T = SmemLSTM; };

__device__ __forceinline__ float sigm(float x) { return 1.0f / (1.0f + expf(-x)); }
__device__ __forceinline__ float gelu_exact(float x) {
    return 0.5f * x * (1.0f + erff(x * 0.7071067811865475f));
}

// ---------------------------------------------------------------- utilities
__global__ void k_cvt16(const float* __restrict__ s, _Float16* __restrict__ d, int n) {
    int i = blockIdx.x * blockDim.x + threadIdx.x;
    if (i < n) d[i] = (_Float16)s[i];
}

__global__ void k_zero(uint32_t* __restrict__ p, int n) {
    int i = blockIdx.x * blockDim.x + threadIdx.x;
    if (i < n) p[i] = 0u;
}

// Pack W (row-major [N, Kd] f32, B = W^T) into WMMA B fragments:
// frag = kt*(N/16)+nt, per-lane 16 contiguous halves matching the 16-bit
// B operand lane layout (lane<16: n=lane, K{0-7,16-23}; lane>=16: K{8-15,24-31}).
// interleave!=0 remaps output column n -> original row (n&3)*(N/4) + (n>>2)
// (quad-interleaves the LSTM gates i,f,g,o per feature).
__global__ void k_pack(const float* __restrict__ W, _Float16* __restrict__ Bp,
                       int N, int Kd, int interleave) {
    int idx = blockIdx.x * blockDim.x + threadIdx.x;
    if (idx >= N * Kd) return;
    int frag = idx >> 9;
    int lane = (idx >> 4) & 31;
    int j    = idx & 15;
    int ntPer = N >> 4;
    int kt = frag / ntPer, nt = frag % ntPer;
    int khalf = lane >> 4, nin = lane & 15;
    int kk = (j < 8) ? (khalf * 8 + j) : (16 + khalf * 8 + (j - 8));
    int k  = kt * 32 + kk;
    int n  = nt * 16 + nin;
    int orig = interleave ? ((n & 3) * (N >> 2) + (n >> 2)) : n;
    Bp[idx] = (_Float16)W[(size_t)orig * Kd + k];
}

__global__ void k_biascomb(const float* __restrict__ bih, const float* __restrict__ bhh,
                           float* __restrict__ bc, int N) {
    int n = blockIdx.x * blockDim.x + threadIdx.x;
    if (n < N) {
        int orig = (n & 3) * (N >> 2) + (n >> 2);
        bc[n] = bih[orig] + bhh[orig];
    }
}

// ---------------------------------------------------------------- WMMA GEMM
// C[16384 x NTOT] = A_cat[16384 x 32*(KT0+KT1)] * Bpacked, f16 in / f32 acc.
// K-tiles [0,KT0) read A0 (optionally row-gathered via idx0/step),
// K-tiles [KT0,KT0+KT1) read A1. All A sources are row-major f16, stride FD.
// A tiles are staged with CDNA5 async global->LDS DMA (ASYNCcnt), double-buffered.
template<int EPI, bool G0>
__global__ __launch_bounds__(256)
void k_gemm(const _Float16* __restrict__ A0, const int* __restrict__ idx0, int step,
            const _Float16* __restrict__ A1,
            const _Float16* __restrict__ B0, const _Float16* __restrict__ B1,
            int NTOT, int KT0, int KT1,
            const float* __restrict__ bias,
            _Float16* __restrict__ outH, float* __restrict__ outC,
            float* __restrict__ outF, _Float16* __restrict__ out16) {
    __shared__ typename SmemSel<EPI>::T sm;
    const int tid  = threadIdx.x;
    const int lane = tid & 31;
    const int wave = tid >> 5;
    const int wm = wave >> 1, wn = wave & 1;
    const int mBase = blockIdx.x * BM;
    const int nBase = blockIdx.y * BN;

    // LDS staging assignment: each thread owns 32B of one A row per K-tile
    const int srow = tid >> 1, shalf = tid & 1;
    const int smRow = mBase + srow;
    const _Float16* a0row;
    if constexpr (G0) a0row = A0 + (size_t)idx0[smRow * KNBR + step] * FD;
    else              a0row = A0 + (size_t)smRow * FD;
    const _Float16* a1row = A1 + (size_t)smRow * FD;

    const uint64_t g0 = (uint64_t)(uintptr_t)a0row + (uint64_t)shalf * 32u;
    const uint64_t g1 = (uint64_t)(uintptr_t)a1row + (uint64_t)shalf * 32u;
    const uint32_t ldsD[2] = {
        (uint32_t)(uintptr_t)&sm.As[0][srow * 32 + shalf * 16],
        (uint32_t)(uintptr_t)&sm.As[1][srow * 32 + shalf * 16]
    };

    const int KT = KT0 + KT1;
    const int ntPer = NTOT >> 4;
    const int khalf = lane >> 4;

    // issue one 32B async copy (2 x b128) for k-tile kt into LDS buffer buf
    auto issueTile = [&](int kt, int buf) {
        const uint64_t ga = (kt < KT0) ? (g0 + (uint64_t)kt * 64u)
                                       : (g1 + (uint64_t)(kt - KT0) * 64u);
        asm volatile("global_load_async_to_lds_b128 %0, %1, off"
                     :: "v"(ldsD[buf]), "v"(ga) : "memory");
        asm volatile("global_load_async_to_lds_b128 %0, %1, off offset:16"
                     :: "v"(ldsD[buf]), "v"(ga) : "memory");
    };

    v8f acc[2][4];
#pragma unroll
    for (int mi = 0; mi < 2; ++mi)
#pragma unroll
        for (int ni = 0; ni < 4; ++ni)
#pragma unroll
            for (int r = 0; r < 8; ++r) acc[mi][ni][r] = 0.0f;

    issueTile(0, 0);

    for (int kt = 0; kt < KT; ++kt) {
        asm volatile("s_wait_asynccnt 0" ::: "memory");
        __syncthreads();                  // tile kt visible to all waves; all
                                          // readers of buffer (kt+1)&1 are done
        if (kt + 1 < KT) issueTile(kt + 1, (kt + 1) & 1);

        const _Float16* As = sm.As[kt & 1];

        // A fragments (16-bit A operand layout, 16x32 per tile)
        v16h af[2];
#pragma unroll
        for (int mi = 0; mi < 2; ++mi) {
            const _Float16* pa = &As[(wm * 32 + mi * 16 + (lane & 15)) * 32 + khalf * 8];
            v8h lo = *(const v8h*)pa;
            v8h hi = *(const v8h*)(pa + 16);
            af[mi] = __builtin_shufflevector(lo, hi, 0, 1, 2, 3, 4, 5, 6, 7,
                                                     8, 9, 10, 11, 12, 13, 14, 15);
        }

        const _Float16* Bp = (kt < KT0) ? B0 : B1;
        const int ktL = (kt < KT0) ? kt : kt - KT0;
        if (kt + 1 < KT) {  // L2 prefetch of the next k-tile's first B fragment
            const _Float16* Bq = (kt + 1 < KT0) ? B0 : B1;
            const int kq = (kt + 1 < KT0) ? kt + 1 : kt + 1 - KT0;
            __builtin_prefetch(Bq + (((size_t)(kq * ntPer + (nBase >> 4) + wn * 4)) << 9) +
                                   (lane << 4), 0, 1);
        }
#pragma unroll
        for (int ni = 0; ni < 4; ++ni) {
            const int nt = (nBase >> 4) + wn * 4 + ni;
            const v16h bf = *(const v16h*)(Bp + (((size_t)(ktL * ntPer + nt)) << 9) + (lane << 4));
#pragma unroll
            for (int mi = 0; mi < 2; ++mi)
                acc[mi][ni] = __builtin_amdgcn_wmma_f32_16x16x32_f16(
                    false, af[mi], false, bf, (short)0, acc[mi][ni], false, false);
        }
        __syncthreads();                  // done reading As[kt&1] before the
                                          // async for kt+2 may overwrite it
    }

    if constexpr (EPI == EPI_LSTM) {
        // Gate columns are quad-interleaved (i,f,g,o per feature). Stage 64-col
        // chunks to LDS, then apply the LSTM cell.
#pragma unroll
        for (int chunk = 0; chunk < 2; ++chunk) {
            __syncthreads();
            if (wn == chunk) {
#pragma unroll
                for (int mi = 0; mi < 2; ++mi)
#pragma unroll
                    for (int ni = 0; ni < 4; ++ni)
#pragma unroll
                        for (int r = 0; r < 8; ++r)
                            sm.gbuf[(wm * 32 + mi * 16 + r + (lane >> 4) * 8) * 64 +
                                    ni * 16 + (lane & 15)] = acc[mi][ni][r];
            }
            __syncthreads();
            const int nCB = nBase + chunk * 64;
            for (int e = tid; e < BM * 16; e += 256) {
                const int row = e >> 4, ff = e & 15;
                const int m  = mBase + row;
                const int fg = (nCB >> 2) + ff;            // global feature index
                const float q0 = sm.gbuf[row * 64 + ff * 4 + 0] + bias[nCB + ff * 4 + 0];
                const float q1 = sm.gbuf[row * 64 + ff * 4 + 1] + bias[nCB + ff * 4 + 1];
                const float q2 = sm.gbuf[row * 64 + ff * 4 + 2] + bias[nCB + ff * 4 + 2];
                const float q3 = sm.gbuf[row * 64 + ff * 4 + 3] + bias[nCB + ff * 4 + 3];
                const size_t o = (size_t)m * FD + fg;
                const float cn = sigm(q1) * outC[o] + sigm(q0) * tanhf(q2);
                outC[o] = cn;
                outH[o] = (_Float16)(sigm(q3) * tanhf(cn));
            }
        }
    } else {
#pragma unroll
        for (int mi = 0; mi < 2; ++mi)
#pragma unroll
            for (int ni = 0; ni < 4; ++ni)
#pragma unroll
                for (int r = 0; r < 8; ++r) {
                    const int m = mBase + wm * 32 + mi * 16 + r + (lane >> 4) * 8;
                    const int n = nBase + wn * 64 + ni * 16 + (lane & 15);
                    float v = gelu_exact(acc[mi][ni][r] + bias[n]);
                    const size_t o = (size_t)m * NTOT + n;
                    if constexpr (EPI == EPI_SAGE_ST)       outF[o] = v;
                    else if constexpr (EPI == EPI_SAGE_ADD) outF[o] += v;
                    else                                    out16[o] = (_Float16)v;
                }
    }
}

// ---------------------------------------------------------------- post ops
__global__ void k_layernorm(float* __restrict__ h, const float* __restrict__ g,
                            const float* __restrict__ b) {
    __shared__ float red[FD];
    const int row = blockIdx.x, c = threadIdx.x;
    const size_t o = (size_t)row * FD + c;
    const float v = h[o];
    red[c] = v; __syncthreads();
    for (int s = FD / 2; s > 0; s >>= 1) { if (c < s) red[c] += red[c + s]; __syncthreads(); }
    const float mu = red[0] * (1.0f / FD);
    __syncthreads();
    const float d = v - mu;
    red[c] = d * d; __syncthreads();
    for (int s = FD / 2; s > 0; s >>= 1) { if (c < s) red[c] += red[c + s]; __syncthreads(); }
    const float var = red[0] * (1.0f / FD);
    h[o] = d * rsqrtf(var + 1e-5f) * g[c] + b[c];
}

__global__ void k_bnstats(const float* __restrict__ h, float* __restrict__ stats) {
    __shared__ float s1[256], s2[256];
    const int col = blockIdx.x, t = threadIdx.x;
    float a = 0.0f, b = 0.0f;
    for (int r = t; r < NR; r += 256) {
        const float v = h[(size_t)r * FD + col];
        a += v; b += v * v;
    }
    s1[t] = a; s2[t] = b; __syncthreads();
    for (int s = 128; s > 0; s >>= 1) {
        if (t < s) { s1[t] += s1[t + s]; s2[t] += s2[t + s]; }
        __syncthreads();
    }
    if (t == 0) {
        const float m = s1[0] * (1.0f / NR);
        stats[col] = m;
        stats[FD + col] = s2[0] * (1.0f / NR) - m * m;
    }
}

__global__ void k_bnapply(const float* __restrict__ h, const float* __restrict__ stats,
                          const float* __restrict__ g, const float* __restrict__ b,
                          _Float16* __restrict__ o) {
    const int i = blockIdx.x * blockDim.x + threadIdx.x;
    const int c = i & (FD - 1);
    const float v = (h[i] - stats[c]) * rsqrtf(stats[FD + c] + 1e-5f) * g[c] + b[c];
    o[i] = (_Float16)v;
}

__global__ void k_head(const _Float16* __restrict__ x, const float* __restrict__ W2,
                       const float* __restrict__ b2, float* __restrict__ out) {
    const int i = blockIdx.x * blockDim.x + threadIdx.x;
    if (i >= NR * 2) return;
    const int row = i >> 1, cls = i & 1;
    const _Float16* xr = x + (size_t)row * FD;
    const float* w = W2 + cls * FD;
    float acc = 0.0f;
    for (int k = 0; k < FD; ++k) acc += (float)xr[k] * w[k];
    out[i] = acc + b2[cls];
}

// ---------------------------------------------------------------- launch
extern "C" void kernel_launch(void* const* d_in, const int* in_sizes, int n_in,
                              void* d_out, int out_size, void* d_ws, size_t ws_size,
                              hipStream_t stream) {
    (void)in_sizes; (void)n_in; (void)out_size; (void)ws_size;
    const float* x_user    = (const float*)d_in[0];
    const float* x_product = (const float*)d_in[1];
    const float* x_review  = (const float*)d_in[2];
    const int*   nbrp[2]   = { (const int*)d_in[3], (const int*)d_in[4] };
    const float* Wih[2]    = { (const float*)d_in[5],  (const float*)d_in[12] };
    const float* Whh[2]    = { (const float*)d_in[6],  (const float*)d_in[13] };
    const float* bih[2]    = { (const float*)d_in[7],  (const float*)d_in[14] };
    const float* bhh[2]    = { (const float*)d_in[8],  (const float*)d_in[15] };
    const float* Wself[2]  = { (const float*)d_in[9],  (const float*)d_in[16] };
    const float* Wnei[2]   = { (const float*)d_in[10], (const float*)d_in[17] };
    const float* bconv[2]  = { (const float*)d_in[11], (const float*)d_in[18] };
    const float* ln_g = (const float*)d_in[19];
    const float* ln_b = (const float*)d_in[20];
    const float* bn_g = (const float*)d_in[21];
    const float* bn_b = (const float*)d_in[22];
    const float* W1   = (const float*)d_in[23];
    const float* b1   = (const float*)d_in[24];
    const float* W2   = (const float*)d_in[25];
    const float* b2   = (const float*)d_in[26];
    float* out = (float*)d_out;

    uint8_t* ws = (uint8_t*)d_ws;
    size_t off = 0;
    auto alloc = [&](size_t bytes) -> void* {
        void* p = ws + off;
        off += (bytes + 255) & ~(size_t)255;
        return p;
    };

    _Float16* xu16 = (_Float16*)alloc((size_t)65536 * FD * 2);
    _Float16* xp16 = (_Float16*)alloc((size_t)65536 * FD * 2);
    _Float16* xr16 = (_Float16*)alloc((size_t)NR * FD * 2);
    _Float16* hA[2], *hB[2]; float* cbuf[2];
    _Float16* WihP[2], *WhhP[2], *WselfP[2], *WneiP[2]; float* bcomb[2];
    for (int r = 0; r < 2; ++r) {
        hA[r]     = (_Float16*)alloc((size_t)NR * FD * 2);
        hB[r]     = (_Float16*)alloc((size_t)NR * FD * 2);
        cbuf[r]   = (float*)alloc((size_t)NR * FD * 4);
        WihP[r]   = (_Float16*)alloc((size_t)1024 * 256 * 2);
        WhhP[r]   = (_Float16*)alloc((size_t)1024 * 256 * 2);
        bcomb[r]  = (float*)alloc(1024 * 4);
        WselfP[r] = (_Float16*)alloc((size_t)256 * 256 * 2);
        WneiP[r]  = (_Float16*)alloc((size_t)256 * 256 * 2);
    }
    _Float16* W1P   = (_Float16*)alloc((size_t)256 * 256 * 2);
    float*    hsum  = (float*)alloc((size_t)NR * FD * 4);
    float*    stats = (float*)alloc(2 * FD * 4);
    _Float16* hbn16 = (_Float16*)alloc((size_t)NR * FD * 2);
    _Float16* mlp16 = (_Float16*)alloc((size_t)NR * FD * 2);

    const int T = 256;
    // f16 conversions of node features
    k_cvt16<<<(65536 * FD + T - 1) / T, T, 0, stream>>>(x_user, xu16, 65536 * FD);
    k_cvt16<<<(65536 * FD + T - 1) / T, T, 0, stream>>>(x_product, xp16, 65536 * FD);
    k_cvt16<<<(NR * FD + T - 1) / T, T, 0, stream>>>(x_review, xr16, NR * FD);
    // weight packing
    for (int r = 0; r < 2; ++r) {
        k_pack<<<(1024 * 256) / T, T, 0, stream>>>(Wih[r], WihP[r], 1024, 256, 1);
        k_pack<<<(1024 * 256) / T, T, 0, stream>>>(Whh[r], WhhP[r], 1024, 256, 1);
        k_biascomb<<<1024 / T, T, 0, stream>>>(bih[r], bhh[r], bcomb[r], 1024);
        k_pack<<<(256 * 256) / T, T, 0, stream>>>(Wself[r], WselfP[r], 256, 256, 0);
        k_pack<<<(256 * 256) / T, T, 0, stream>>>(Wnei[r], WneiP[r], 256, 256, 0);
    }
    k_pack<<<(256 * 256) / T, T, 0, stream>>>(W1, W1P, 256, 256, 0);

    // LSTM recurrence per relation: gates = [x_t | h] @ [Wih ; Whh]^T + b
    const _Float16* xs[2] = { xu16, xp16 };
    for (int r = 0; r < 2; ++r) {
        k_zero<<<(NR * FD / 2) / T, T, 0, stream>>>((uint32_t*)hA[r], NR * FD / 2);
        k_zero<<<(NR * FD) / T, T, 0, stream>>>((uint32_t*)cbuf[r], NR * FD);
        for (int t = 0; t < KNBR; ++t) {
            _Float16* hin  = (t & 1) ? hB[r] : hA[r];
            _Float16* hout = (t & 1) ? hA[r] : hB[r];
            k_gemm<EPI_LSTM, true><<<dim3(NR / BM, 1024 / BN), T, 0, stream>>>(
                xs[r], nbrp[r], t, hin, WihP[r], WhhP[r], 1024, 8, 8,
                bcomb[r], hout, cbuf[r], nullptr, nullptr);
        }
        // after 16 steps the final hidden state is in hA[r]
    }

    // SAGE: gelu(x_review@Wself^T + h@Wneigh^T + bconv), summed over relations
    k_gemm<EPI_SAGE_ST, false><<<dim3(NR / BM, 256 / BN), T, 0, stream>>>(
        xr16, nullptr, 0, hA[0], WselfP[0], WneiP[0], 256, 8, 8,
        bconv[0], nullptr, nullptr, hsum, nullptr);
    k_gemm<EPI_SAGE_ADD, false><<<dim3(NR / BM, 256 / BN), T, 0, stream>>>(
        xr16, nullptr, 0, hA[1], WselfP[1], WneiP[1], 256, 8, 8,
        bconv[1], nullptr, nullptr, hsum, nullptr);

    // LayerNorm (per row), BatchNorm (batch stats) -> f16
    k_layernorm<<<NR, FD, 0, stream>>>(hsum, ln_g, ln_b);
    k_bnstats<<<FD, 256, 0, stream>>>(hsum, stats);
    k_bnapply<<<(NR * FD) / T, T, 0, stream>>>(hsum, stats, bn_g, bn_b, hbn16);

    // MLP: gelu(h@W1^T + b1) via WMMA, then [2,256] head
    k_gemm<EPI_GELU16, false><<<dim3(NR / BM, 256 / BN), T, 0, stream>>>(
        hbn16, nullptr, 0, hbn16, W1P, nullptr, 256, 8, 0,
        b1, nullptr, nullptr, nullptr, mlp16);
    k_head<<<(NR * 2 + T - 1) / T, T, 0, stream>>>(mlp16, W2, b2, out);
}
